// EnhancedTransformerLayer_81063212744882
// MI455X (gfx1250) — compile-verified, matrix operand order checked by reading, not run on hardware
//
#include <hip/hip_runtime.h>

// ---------------------------------------------------------------------------
// Types and helpers
// ---------------------------------------------------------------------------
typedef unsigned short u16;
typedef __attribute__((ext_vector_type(16))) __bf16 v16bf;
typedef __attribute__((ext_vector_type(8)))  float  v8f;

struct U128 { unsigned int x, y, z, w; };
union Frag { v16bf v; U128 q[2]; };

static __device__ __forceinline__ u16 f32_to_bf16(float f) {
  unsigned int u = __float_as_uint(f);
  unsigned int r = (u + 0x7FFFu + ((u >> 16) & 1u)) >> 16;
  return (u16)r;
}

static __device__ __forceinline__ v8f vzero8() {
  v8f z = {0.f, 0.f, 0.f, 0.f, 0.f, 0.f, 0.f, 0.f};
  return z;
}

static __device__ __forceinline__ v8f wmma_bf16(const Frag& a, const Frag& b, v8f c) {
  return __builtin_amdgcn_wmma_f32_16x16x32_bf16(
      false, a.v, false, b.v, (short)0, c, false, false);
}

// --- Async global->LDS staging (CDNA5 GLOBAL_LOAD_ASYNC_TO_LDS_B128, ASYNCcnt)
#if defined(__has_builtin)
#if __has_builtin(__builtin_amdgcn_global_load_async_to_lds_b128) && \
    __has_builtin(__builtin_amdgcn_s_wait_asynccnt)
#define HAVE_ASYNC_LDS 1
#endif
#endif

#ifdef HAVE_ASYNC_LDS
typedef int v4i_t __attribute__((vector_size(16)));
typedef __attribute__((address_space(1))) v4i_t* gv4i_p;  // global src (VADDR)
typedef __attribute__((address_space(3))) v4i_t* lv4i_p;  // LDS dst (VDST)
static __device__ __forceinline__ void async_cp16(void* lds, const void* g) {
  __builtin_amdgcn_global_load_async_to_lds_b128((gv4i_p)g, (lv4i_p)lds, 0, 0);
}
static __device__ __forceinline__ void wait_async0() {
  __builtin_amdgcn_s_wait_asynccnt(0);
}
#endif

// Problem constants
#define DM   1024
#define FF   4096
#define BB   8
#define SS   1024
#define NH   16
#define DK   64
#define NTOK (BB * SS)   // 8192

// LDS row widths (u16 units), padded for bank-conflict-free b128 fragment reads
#define GLW  40   // gemm tiles: 80B rows (16B-aligned, 20-bank stride)
#define FLW  72   // flash tiles: 144B rows (16B-aligned, 36-bank stride)

// ---------------------------------------------------------------------------
// fp32 -> bf16 conversion (plain)
// ---------------------------------------------------------------------------
__global__ void cvt_bf16_kernel(const float* __restrict__ in, u16* __restrict__ out, int n) {
  int i = blockIdx.x * 256 + threadIdx.x;
  if (i < n) out[i] = f32_to_bf16(in[i]);
}

// fp32 [K][N] -> bf16 [N][K] (weight transpose, done once)
__global__ __launch_bounds__(256) void cvt_transpose_bf16_kernel(
    const float* __restrict__ in, u16* __restrict__ out, int K, int N) {
  __shared__ float t[32][33];
  const int tx = threadIdx.x & 31, ty = threadIdx.x >> 5;
  const int kb = blockIdx.y * 32, nb = blockIdx.x * 32;
#pragma unroll
  for (int i = 0; i < 4; i++)
    t[ty + 8 * i][tx] = in[(size_t)(kb + ty + 8 * i) * N + nb + tx];
  __syncthreads();
#pragma unroll
  for (int i = 0; i < 4; i++)
    out[(size_t)(nb + ty + 8 * i) * K + kb + tx] = f32_to_bf16(t[tx][ty + 8 * i]);
}

// bf16 [B][S][DM] -> bf16 [B][DM][S] (V transpose for flash B-fragments)
__global__ __launch_bounds__(256) void transpose_bf16_kernel(
    const u16* __restrict__ in, u16* __restrict__ out) {
  __shared__ u16 t[32][33];
  const int tx = threadIdx.x & 31, ty = threadIdx.x >> 5;
  const int c0 = blockIdx.x * 32;  // DM dim
  const int r0 = blockIdx.y * 32;  // S dim
  const size_t ib = (size_t)blockIdx.z * SS * DM;
  const size_t ob = (size_t)blockIdx.z * DM * SS;
#pragma unroll
  for (int i = 0; i < 4; i++)
    t[ty + 8 * i][tx] = in[ib + (size_t)(r0 + ty + 8 * i) * DM + c0 + tx];
  __syncthreads();
#pragma unroll
  for (int i = 0; i < 4; i++)
    out[ob + (size_t)(c0 + ty + 8 * i) * SS + r0 + tx] = t[tx][ty + 8 * i];
}

// ---------------------------------------------------------------------------
// Tiled WMMA GEMM: C[M,N] = act(A[M,K] @ B[K,N] + bias), B given transposed
// (Bt[N][K]).  Block tile 128x128 (8 waves x 32x64), BK=32, double-buffered.
// Staging uses GLOBAL_LOAD_ASYNC_TO_LDS_B128 when available.
// OUT_BF16: 0 -> fp32 out, 1 -> bf16 out.  ACT: 0=none, 1=gelu, 2=sigmoid
// ---------------------------------------------------------------------------
template <int OUT_BF16, int ACT>
__global__ __launch_bounds__(256) void gemm_kernel(
    const u16* __restrict__ Ag, const u16* __restrict__ Bt,
    const float* __restrict__ bias, float* __restrict__ Cf, u16* __restrict__ Ch,
    int K, int N) {
  __shared__ u16 As[2][128][GLW];
  __shared__ u16 Bs[2][128][GLW];

  const int tid  = threadIdx.x;
  const int lane = tid & 31;
  const int wv   = tid >> 5;
  const int wm   = (wv & 3) * 32;   // wave row offset
  const int wn   = (wv >> 2) * 64;  // wave col offset
  const int ln   = lane & 15;
  const int kh   = (lane < 16) ? 0 : 8;

  const size_t rowBase = (size_t)blockIdx.y * 128;
  const size_t colBase = (size_t)blockIdx.x * 128;

  v8f acc[2][4];
#pragma unroll
  for (int i = 0; i < 2; i++)
#pragma unroll
    for (int j = 0; j < 4; j++) acc[i][j] = vzero8();

  const int lr = tid >> 2, lc = (tid & 3) * 8;  // loader: 64 rows/pass x 4 chunks
  const u16* apB = Ag + (rowBase + lr) * (size_t)K + lc;
  const u16* bpB = Bt + (colBase + lr) * (size_t)K + lc;

  // Preload tile 0
#ifdef HAVE_ASYNC_LDS
  async_cp16(&As[0][lr][lc],      apB);
  async_cp16(&As[0][lr + 64][lc], apB + (size_t)64 * K);
  async_cp16(&Bs[0][lr][lc],      bpB);
  async_cp16(&Bs[0][lr + 64][lc], bpB + (size_t)64 * K);
  wait_async0();
#else
  *(U128*)&As[0][lr][lc]      = *(const U128*)apB;
  *(U128*)&As[0][lr + 64][lc] = *(const U128*)(apB + (size_t)64 * K);
  *(U128*)&Bs[0][lr][lc]      = *(const U128*)bpB;
  *(U128*)&Bs[0][lr + 64][lc] = *(const U128*)(bpB + (size_t)64 * K);
#endif
  __syncthreads();

  int cur = 0;
  for (int kb = 0; kb < K; kb += 32) {
    const bool hasNext = (kb + 32 < K);
    const int nxt = cur ^ 1;
#ifdef HAVE_ASYNC_LDS
    if (hasNext) {  // issue async stage of next tile; overlaps with WMMAs below
      async_cp16(&As[nxt][lr][lc],      apB + kb + 32);
      async_cp16(&As[nxt][lr + 64][lc], apB + (size_t)64 * K + kb + 32);
      async_cp16(&Bs[nxt][lr][lc],      bpB + kb + 32);
      async_cp16(&Bs[nxt][lr + 64][lc], bpB + (size_t)64 * K + kb + 32);
    }
#else
    U128 pa0 = {}, pa1 = {}, pb0 = {}, pb1 = {};
    if (hasNext) {
      pa0 = *(const U128*)(apB + kb + 32);
      pa1 = *(const U128*)(apB + (size_t)64 * K + kb + 32);
      pb0 = *(const U128*)(bpB + kb + 32);
      pb1 = *(const U128*)(bpB + (size_t)64 * K + kb + 32);
    }
#endif

    Frag a0, a1;
    a0.q[0] = *(const U128*)&As[cur][wm + ln][kh];       a0.q[1] = *(const U128*)&As[cur][wm + ln][kh + 16];
    a1.q[0] = *(const U128*)&As[cur][wm + 16 + ln][kh];  a1.q[1] = *(const U128*)&As[cur][wm + 16 + ln][kh + 16];
#pragma unroll
    for (int j = 0; j < 4; j++) {
      Frag b;
      const int bc = wn + j * 16 + ln;
      b.q[0] = *(const U128*)&Bs[cur][bc][kh];
      b.q[1] = *(const U128*)&Bs[cur][bc][kh + 16];
      acc[0][j] = wmma_bf16(a0, b, acc[0][j]);
      acc[1][j] = wmma_bf16(a1, b, acc[1][j]);
    }

#ifdef HAVE_ASYNC_LDS
    if (hasNext) wait_async0();   // own-wave async copies must land before barrier
#else
    if (hasNext) {
      *(U128*)&As[nxt][lr][lc]      = pa0;
      *(U128*)&As[nxt][lr + 64][lc] = pa1;
      *(U128*)&Bs[nxt][lr][lc]      = pb0;
      *(U128*)&Bs[nxt][lr + 64][lc] = pb1;
    }
#endif
    __syncthreads();
    cur ^= 1;
  }

  const int rh = (lane < 16) ? 0 : 8;
#pragma unroll
  for (int i = 0; i < 2; i++) {
#pragma unroll
    for (int j = 0; j < 4; j++) {
      v8f a = acc[i][j];
      const size_t col = colBase + wn + j * 16 + ln;
      const float bia = bias[col];
#pragma unroll
      for (int r = 0; r < 8; r++) {
        const size_t row = rowBase + wm + i * 16 + rh + r;
        float v = a[r] + bia;
        if (ACT == 1) v = 0.5f * v * (1.0f + erff(v * 0.70710678118654752f));
        else if (ACT == 2) v = 1.0f / (1.0f + __expf(-v));
        if (OUT_BF16) Ch[row * (size_t)N + col] = f32_to_bf16(v);
        else          Cf[row * (size_t)N + col] = v;
      }
    }
  }
}

// ---------------------------------------------------------------------------
// Flash attention. Grid: (S/128, NH, B). Block: 256 (8 waves, 16 q-rows each).
// Q/K: bf16 [B,S,DM] (head h at cols h*64..).  Vt: bf16 [B,DM,S] (pre-transposed).
// ---------------------------------------------------------------------------
__global__ __launch_bounds__(256) void flash_attn_kernel(
    const u16* __restrict__ Qg, const u16* __restrict__ Kg,
    const u16* __restrict__ Vtg, u16* __restrict__ ctx) {
  __shared__ u16 Qs[128][FLW];    // 18 KB
  __shared__ u16 Ks[64][FLW];     //  9 KB  (Ks[key][d])
  __shared__ u16 Vt[64][FLW];     //  9 KB  (Vt[d][key])
  __shared__ u16 Ps[8][16][FLW];  // 18 KB  per-wave P tile

  const int tid  = threadIdx.x;
  const int lane = tid & 31;
  const int wq   = tid >> 5;
  const int ln   = lane & 15;
  const int kh   = (lane < 16) ? 0 : 8;

  const int b = blockIdx.z, h = blockIdx.y;
  const int qbase = blockIdx.x * 128;
  const size_t base  = (size_t)b * SS * DM + (size_t)h * DK;      // + s*DM + d
  const size_t vbase = ((size_t)b * DM + (size_t)h * DK) * SS;    // + d*SS + s

  // Load Q tile (128 x 64)
#pragma unroll
  for (int i = 0; i < 4; i++) {
    int idx = tid + i * 256;
    int r = idx >> 3, c = (idx & 7) * 8;
    *(U128*)&Qs[r][c] = *(const U128*)(Qg + base + (size_t)(qbase + r) * DM + c);
  }

  float m_[8], l_[8];
  v8f O[4];
#pragma unroll
  for (int r = 0; r < 8; r++) { m_[r] = -1e30f; l_[r] = 0.f; }
#pragma unroll
  for (int t = 0; t < 4; t++) O[t] = vzero8();

  __syncthreads();

  for (int kt = 0; kt < SS / 64; kt++) {
    const int kb = kt * 64;
    // Stage K (natural) and Vt (pre-transposed) tiles
#pragma unroll
    for (int i = 0; i < 2; i++) {
      int idx = tid + i * 256;
      int r = idx >> 3, c = (idx & 7) * 8;
#ifdef HAVE_ASYNC_LDS
      async_cp16(&Ks[r][c], Kg + base + (size_t)(kb + r) * DM + c);
      async_cp16(&Vt[r][c], Vtg + vbase + (size_t)r * SS + kb + c);
#else
      *(U128*)&Ks[r][c] = *(const U128*)(Kg + base + (size_t)(kb + r) * DM + c);
      *(U128*)&Vt[r][c] = *(const U128*)(Vtg + vbase + (size_t)r * SS + kb + c);
#endif
    }
#ifdef HAVE_ASYNC_LDS
    wait_async0();
#endif
    __syncthreads();

    // S = Q @ K^T
    v8f sa[4];
#pragma unroll
    for (int t = 0; t < 4; t++) sa[t] = vzero8();

    Frag a0, a1;
    const int qr = wq * 16 + ln;
    a0.q[0] = *(const U128*)&Qs[qr][kh];       a0.q[1] = *(const U128*)&Qs[qr][kh + 16];
    a1.q[0] = *(const U128*)&Qs[qr][32 + kh];  a1.q[1] = *(const U128*)&Qs[qr][48 + kh];
#pragma unroll
    for (int t = 0; t < 4; t++) {
      const int kc = t * 16 + ln;
      Frag b0, b1;
      b0.q[0] = *(const U128*)&Ks[kc][kh];       b0.q[1] = *(const U128*)&Ks[kc][kh + 16];
      b1.q[0] = *(const U128*)&Ks[kc][32 + kh];  b1.q[1] = *(const U128*)&Ks[kc][48 + kh];
      sa[t] = wmma_bf16(a0, b0, sa[t]);
      sa[t] = wmma_bf16(a1, b1, sa[t]);
    }

    // Online softmax update (row = (lane<16 ? r : 8+r) within wave tile)
#pragma unroll
    for (int r = 0; r < 8; r++) {
      float mx = fmaxf(fmaxf(sa[0][r], sa[1][r]), fmaxf(sa[2][r], sa[3][r])) * 0.125f;
#pragma unroll
      for (int msk = 1; msk < 16; msk <<= 1) mx = fmaxf(mx, __shfl_xor(mx, msk, 32));
      float mnew  = fmaxf(m_[r], mx);
      float alpha = __expf(m_[r] - mnew);
      const int rowi = (lane < 16) ? r : (8 + r);
      float rs = 0.f;
#pragma unroll
      for (int t = 0; t < 4; t++) {
        float p = __expf(sa[t][r] * 0.125f - mnew);
        rs += p;
        Ps[wq][rowi][t * 16 + ln] = f32_to_bf16(p);
      }
#pragma unroll
      for (int msk = 1; msk < 16; msk <<= 1) rs += __shfl_xor(rs, msk, 32);
      l_[r] = l_[r] * alpha + rs;
      m_[r] = mnew;
#pragma unroll
      for (int t = 0; t < 4; t++) O[t][r] *= alpha;
    }

    // O += P @ V
    Frag p0, p1;
    p0.q[0] = *(const U128*)&Ps[wq][ln][kh];       p0.q[1] = *(const U128*)&Ps[wq][ln][kh + 16];
    p1.q[0] = *(const U128*)&Ps[wq][ln][32 + kh];  p1.q[1] = *(const U128*)&Ps[wq][ln][48 + kh];
#pragma unroll
    for (int t = 0; t < 4; t++) {
      const int dc = t * 16 + ln;
      Frag v0, v1;
      v0.q[0] = *(const U128*)&Vt[dc][kh];       v0.q[1] = *(const U128*)&Vt[dc][kh + 16];
      v1.q[0] = *(const U128*)&Vt[dc][32 + kh];  v1.q[1] = *(const U128*)&Vt[dc][48 + kh];
      O[t] = wmma_bf16(p0, v0, O[t]);
      O[t] = wmma_bf16(p1, v1, O[t]);
    }
    __syncthreads();
  }

  // Normalize and write context
#pragma unroll
  for (int t = 0; t < 4; t++) {
#pragma unroll
    for (int r = 0; r < 8; r++) {
      const int rowi = (lane < 16) ? r : (8 + r);
      const int s = qbase + wq * 16 + rowi;
      float val = O[t][r] / l_[r];
      ctx[base + (size_t)s * DM + t * 16 + ln] = f32_to_bf16(val);
    }
  }
}

// ---------------------------------------------------------------------------
// Fused depthwise-conv + residual + LayerNorm
// ---------------------------------------------------------------------------
__global__ __launch_bounds__(256) void ln_attn_kernel(
    const float* __restrict__ x, const float* __restrict__ glob,
    const float* __restrict__ cw, const float* __restrict__ cb,
    const float* __restrict__ g, const float* __restrict__ be,
    float* __restrict__ out) {
  const int row = blockIdx.x;
  const int s   = row & (SS - 1);
  const int tid = threadIdx.x;
  const float* xr = x + (size_t)row * DM;
  const float* gr = glob + (size_t)row * DM;

  float y[4]; float sum = 0.f, sq = 0.f;
#pragma unroll
  for (int i = 0; i < 4; i++) {
    int d = tid + i * 256;
    float xv = xr[d];
    float xl = (s > 0)      ? xr[d - DM] : 0.f;
    float xn = (s < SS - 1) ? xr[d + DM] : 0.f;
    float lf = xl * cw[3 * d] + xv * cw[3 * d + 1] + xn * cw[3 * d + 2] + cb[d];
    float v = xv + gr[d] + 0.3f * lf;
    y[i] = v; sum += v; sq += v * v;
  }
  __shared__ float rs[256], rq[256];
  rs[tid] = sum; rq[tid] = sq; __syncthreads();
  for (int o = 128; o > 0; o >>= 1) {
    if (tid < o) { rs[tid] += rs[tid + o]; rq[tid] += rq[tid + o]; }
    __syncthreads();
  }
  float mean = rs[0] * (1.f / DM);
  float var  = rq[0] * (1.f / DM) - mean * mean;
  float inv  = rsqrtf(var + 1e-5f);
  float* orow = out + (size_t)row * DM;
#pragma unroll
  for (int i = 0; i < 4; i++) {
    int d = tid + i * 256;
    orow[d] = (y[i] - mean) * inv * g[d] + be[d];
  }
}

// out = LN(a + b); also emits bf16 copy for downstream GEMMs
__global__ __launch_bounds__(256) void ln_res_kernel(
    const float* __restrict__ a, const float* __restrict__ bsrc,
    const float* __restrict__ g, const float* __restrict__ be,
    float* __restrict__ outf, u16* __restrict__ outh) {
  const int row = blockIdx.x, tid = threadIdx.x;
  const float* ar = a + (size_t)row * DM;
  const float* br = bsrc + (size_t)row * DM;
  float y[4]; float sum = 0.f, sq = 0.f;
#pragma unroll
  for (int i = 0; i < 4; i++) {
    int d = tid + i * 256;
    float v = ar[d] + br[d];
    y[i] = v; sum += v; sq += v * v;
  }
  __shared__ float rs[256], rq[256];
  rs[tid] = sum; rq[tid] = sq; __syncthreads();
  for (int o = 128; o > 0; o >>= 1) {
    if (tid < o) { rs[tid] += rs[tid + o]; rq[tid] += rq[tid + o]; }
    __syncthreads();
  }
  float mean = rs[0] * (1.f / DM);
  float var  = rq[0] * (1.f / DM) - mean * mean;
  float inv  = rsqrtf(var + 1e-5f);
#pragma unroll
  for (int i = 0; i < 4; i++) {
    int d = tid + i * 256;
    float v = (y[i] - mean) * inv * g[d] + be[d];
    outf[(size_t)row * DM + d] = v;
    outh[(size_t)row * DM + d] = f32_to_bf16(v);
  }
}

// out = LN(h + f3 * gate)
__global__ __launch_bounds__(256) void ln_final_kernel(
    const float* __restrict__ h, const float* __restrict__ f3,
    const float* __restrict__ gate, const float* __restrict__ g,
    const float* __restrict__ be, float* __restrict__ out) {
  const int row = blockIdx.x, tid = threadIdx.x;
  const size_t roff = (size_t)row * DM;
  float y[4]; float sum = 0.f, sq = 0.f;
#pragma unroll
  for (int i = 0; i < 4; i++) {
    int d = tid + i * 256;
    float v = h[roff + d] + f3[roff + d] * gate[roff + d];
    y[i] = v; sum += v; sq += v * v;
  }
  __shared__ float rs[256], rq[256];
  rs[tid] = sum; rq[tid] = sq; __syncthreads();
  for (int o = 128; o > 0; o >>= 1) {
    if (tid < o) { rs[tid] += rs[tid + o]; rq[tid] += rq[tid + o]; }
    __syncthreads();
  }
  float mean = rs[0] * (1.f / DM);
  float var  = rq[0] * (1.f / DM) - mean * mean;
  float inv  = rsqrtf(var + 1e-5f);
#pragma unroll
  for (int i = 0; i < 4; i++) {
    int d = tid + i * 256;
    out[roff + d] = (y[i] - mean) * inv * g[d] + be[d];
  }
}

// ---------------------------------------------------------------------------
// Host launcher
// ---------------------------------------------------------------------------
extern "C" void kernel_launch(void* const* d_in, const int* in_sizes, int n_in,
                              void* d_out, int out_size, void* d_ws, size_t ws_size,
                              hipStream_t stream) {
  (void)in_sizes; (void)n_in; (void)out_size; (void)ws_size;

  const float* x      = (const float*)d_in[0];
  const float* Wq     = (const float*)d_in[1];
  const float* bq     = (const float*)d_in[2];
  const float* Wk     = (const float*)d_in[3];
  const float* bk     = (const float*)d_in[4];
  const float* Wv     = (const float*)d_in[5];
  const float* bv     = (const float*)d_in[6];
  const float* Wo     = (const float*)d_in[7];
  const float* bo     = (const float*)d_in[8];
  const float* conv_w = (const float*)d_in[9];
  const float* conv_b = (const float*)d_in[10];
  const float* lna_g  = (const float*)d_in[11];
  const float* lna_b  = (const float*)d_in[12];
  const float* W1     = (const float*)d_in[13];
  const float* b1     = (const float*)d_in[14];
  const float* W2     = (const float*)d_in[15];
  const float* b2     = (const float*)d_in[16];
  const float* W3     = (const float*)d_in[17];
  const float* b3     = (const float*)d_in[18];
  const float* Wg     = (const float*)d_in[19];
  const float* bg     = (const float*)d_in[20];
  const float* n1_g   = (const float*)d_in[21];
  const float* n1_b   = (const float*)d_in[22];
  const float* n2_g   = (const float*)d_in[23];
  const float* n2_b   = (const float*)d_in[24];
  float* out = (float*)d_out;

  // Workspace layout
  char* ws = (char*)d_ws;
  size_t off = 0;
  auto wsp = [&](size_t bytes) -> char* {
    char* p = ws + off;
    off += (bytes + 255) & ~(size_t)255;
    return p;
  };
  u16*   xb   = (u16*)wsp((size_t)NTOK * DM * 2);
  u16*   wqt  = (u16*)wsp((size_t)DM * DM * 2);   // transposed bf16 weights [N][K]
  u16*   wkt  = (u16*)wsp((size_t)DM * DM * 2);
  u16*   wvt  = (u16*)wsp((size_t)DM * DM * 2);
  u16*   wot  = (u16*)wsp((size_t)DM * DM * 2);
  u16*   wgt  = (u16*)wsp((size_t)DM * DM * 2);
  u16*   w1t  = (u16*)wsp((size_t)DM * FF * 2);
  u16*   w2t  = (u16*)wsp((size_t)FF * FF * 2);
  u16*   w3t  = (u16*)wsp((size_t)FF * DM * 2);
  u16*   qb   = (u16*)wsp((size_t)NTOK * DM * 2);
  u16*   kbuf = (u16*)wsp((size_t)NTOK * DM * 2);
  u16*   vbuf = (u16*)wsp((size_t)NTOK * DM * 2);
  u16*   vtb  = (u16*)wsp((size_t)NTOK * DM * 2); // V transposed [B][DM][S]
  u16*   ctxb = (u16*)wsp((size_t)NTOK * DM * 2);
  float* glob = (float*)wsp((size_t)NTOK * DM * 4);
  float* aout = (float*)wsp((size_t)NTOK * DM * 4);
  float* hbuf = (float*)wsp((size_t)NTOK * DM * 4);
  u16*   hb   = (u16*)wsp((size_t)NTOK * DM * 2);
  u16*   f1b  = (u16*)wsp((size_t)NTOK * FF * 2);
  u16*   f2b  = (u16*)wsp((size_t)NTOK * FF * 2);
  float* f3   = (float*)wsp((size_t)NTOK * DM * 4);
  float* gate = (float*)wsp((size_t)NTOK * DM * 4);

  dim3 blk(256);
  auto cvt = [&](const float* src, u16* dst, size_t n) {
    cvt_bf16_kernel<<<dim3((unsigned)((n + 255) / 256)), blk, 0, stream>>>(src, dst, (int)n);
  };
  auto cvtT = [&](const float* src, u16* dst, int K_, int N_) {
    cvt_transpose_bf16_kernel<<<dim3(N_ / 32, K_ / 32), blk, 0, stream>>>(src, dst, K_, N_);
  };

  // 1) conversions (weights transposed once -> GEMM B staging is pure b128)
  cvt(x, xb, (size_t)NTOK * DM);
  cvtT(Wq, wqt, DM, DM);
  cvtT(Wk, wkt, DM, DM);
  cvtT(Wv, wvt, DM, DM);
  cvtT(Wo, wot, DM, DM);
  cvtT(Wg, wgt, DM, DM);
  cvtT(W1, w1t, DM, FF);
  cvtT(W2, w2t, FF, FF);
  cvtT(W3, w3t, FF, DM);

  // 2) Q, K, V projections (bf16 out)
  gemm_kernel<1, 0><<<dim3(DM / 128, NTOK / 128), blk, 0, stream>>>(xb, wqt, bq, nullptr, qb,   DM, DM);
  gemm_kernel<1, 0><<<dim3(DM / 128, NTOK / 128), blk, 0, stream>>>(xb, wkt, bk, nullptr, kbuf, DM, DM);
  gemm_kernel<1, 0><<<dim3(DM / 128, NTOK / 128), blk, 0, stream>>>(xb, wvt, bv, nullptr, vbuf, DM, DM);

  // 3) V transpose, then flash attention -> context (bf16)
  transpose_bf16_kernel<<<dim3(DM / 32, SS / 32, BB), blk, 0, stream>>>(vbuf, vtb);
  flash_attn_kernel<<<dim3(SS / 128, NH, BB), blk, 0, stream>>>(qb, kbuf, vtb, ctxb);

  // 4) Output projection (fp32 out)
  gemm_kernel<0, 0><<<dim3(DM / 128, NTOK / 128), blk, 0, stream>>>(ctxb, wot, bo, glob, nullptr, DM, DM);

  // 5) attn_out = LN(x + glob + 0.3*conv(x))
  ln_attn_kernel<<<dim3(NTOK), blk, 0, stream>>>(x, glob, conv_w, conv_b, lna_g, lna_b, aout);

  // 6) h = LN(x + attn_out), plus bf16 copy
  ln_res_kernel<<<dim3(NTOK), blk, 0, stream>>>(x, aout, n1_g, n1_b, hbuf, hb);

  // 7) FFN chain
  gemm_kernel<1, 1><<<dim3(FF / 128, NTOK / 128), blk, 0, stream>>>(hb,  w1t, b1, nullptr, f1b, DM, FF);
  gemm_kernel<1, 1><<<dim3(FF / 128, NTOK / 128), blk, 0, stream>>>(f1b, w2t, b2, nullptr, f2b, FF, FF);
  gemm_kernel<0, 0><<<dim3(DM / 128, NTOK / 128), blk, 0, stream>>>(f2b, w3t, b3, f3, nullptr, FF, DM);
  gemm_kernel<0, 2><<<dim3(DM / 128, NTOK / 128), blk, 0, stream>>>(hb,  wgt, bg, gate, nullptr, DM, DM);

  // 8) out = LN(h + ffn * gate)
  ln_final_kernel<<<dim3(NTOK), blk, 0, stream>>>(hbuf, f3, gate, n2_g, n2_b, out);
}
